// UpdateONNX_78726750535846
// MI455X (gfx1250) — matrix operand order repaired and checked by reading, not compiled, not run on hardware
//
#include <hip/hip_runtime.h>
#include <hip/hip_bf16.h>

typedef __attribute__((ext_vector_type(16))) _Float16 v16h;
typedef __attribute__((ext_vector_type(8)))  _Float16 v8h;
typedef __attribute__((ext_vector_type(8)))  float    v8f;

#define TILE_M 64
#define TILE_N 128
#define KSTEP  32
#define LDSW   40   // halves per row (80B stride: 16B-aligned, bank-skewed)

#define USE_ASYNC_LDS 1

// ===========================================================================
// WMMA GEMM: Y[M,384] = epi( X[M,K] @ W[384,K]^T + bias ) [+ Y]
//   M % 64 == 0, K % 32 == 0 (ld of X = K), Nout = 384.
// One wave = 16x64 output strip via 4x v_wmma_f32_16x16x32_f16.
// A: global_load_b128 x2 -> cvt f16 -> ds_store_b128.
// B (already f16): global_load_async_to_lds_b128 (ASYNCcnt).
// ===========================================================================
template<int EPI, bool GATHER, bool ADDY, bool RELUX>
__global__ __launch_bounds__(256)
void gemm_wmma_fast(const float* __restrict__ X,
                    const _Float16* __restrict__ W,    // (384, K) f16
                    const float* __restrict__ bias,
                    float* __restrict__ Y,
                    int M, int K, const int* __restrict__ gather)
{
    constexpr int NOUT = 384;
    __shared__ __align__(16) _Float16 As[TILE_M * LDSW];
    __shared__ __align__(16) _Float16 Bs[TILE_N * LDSW];

    const int tid  = threadIdx.x;
    const int lane = tid & 31;
    const int wid  = tid >> 5;
    const int wr   = wid & 3;
    const int wc   = wid >> 2;
    const int rowbase = blockIdx.x * TILE_M;
    const int colbase = blockIdx.y * TILE_N;
    const int mrow = lane & 15;
    const int hsel = lane >> 4;

    v8f acc[4];
    #pragma unroll
    for (int t = 0; t < 4; ++t)
        #pragma unroll
        for (int j = 0; j < 8; ++j) acc[t][j] = 0.0f;

    // A staging: 64 rows x 32 k, 8 floats/thread
    const int arow = tid >> 2;
    const int acol = (tid & 3) * 8;
    const int grow = rowbase + arow;
    const int asrc = GATHER ? gather[grow] : grow;
    const float* aptr = X + (size_t)asrc * K + acol;
    _Float16* asl = &As[arow * LDSW + acol];

    // B staging: 128 cols x 32 k, 16 halves/thread
    const int bcol = tid >> 1;
    const int bk0  = (tid & 1) * 16;
    const _Float16* bptr = W + (size_t)(colbase + bcol) * K + bk0;
    _Float16* bsl = &Bs[bcol * LDSW + bk0];
#if USE_ASYNC_LDS
    const unsigned bs_lds = (unsigned)(size_t)(void*)bsl;
#endif

    for (int k0 = 0; k0 < K; k0 += KSTEP) {
#if USE_ASYNC_LDS
        {
            unsigned long long g0 = (unsigned long long)(size_t)(bptr + k0);
            unsigned long long g1 = (unsigned long long)(size_t)(bptr + k0 + 8);
            asm volatile("global_load_async_to_lds_b128 %0, %1, off"
                         :: "v"(bs_lds), "v"(g0) : "memory");
            asm volatile("global_load_async_to_lds_b128 %0, %1, off"
                         :: "v"(bs_lds + 16), "v"(g1) : "memory");
        }
#else
        *(v8h*)(bsl)     = *(const v8h*)(bptr + k0);
        *(v8h*)(bsl + 8) = *(const v8h*)(bptr + k0 + 8);
#endif
        // A: two 16B loads, pack to 8 halves, one 16B LDS store
        {
            float4 f0 = *(const float4*)(aptr + k0);
            float4 f1 = *(const float4*)(aptr + k0 + 4);
            if (RELUX) {
                f0.x = fmaxf(f0.x, 0.f); f0.y = fmaxf(f0.y, 0.f);
                f0.z = fmaxf(f0.z, 0.f); f0.w = fmaxf(f0.w, 0.f);
                f1.x = fmaxf(f1.x, 0.f); f1.y = fmaxf(f1.y, 0.f);
                f1.z = fmaxf(f1.z, 0.f); f1.w = fmaxf(f1.w, 0.f);
            }
            v8h hv;
            hv[0] = (_Float16)f0.x; hv[1] = (_Float16)f0.y;
            hv[2] = (_Float16)f0.z; hv[3] = (_Float16)f0.w;
            hv[4] = (_Float16)f1.x; hv[5] = (_Float16)f1.y;
            hv[6] = (_Float16)f1.z; hv[7] = (_Float16)f1.w;
            *(v8h*)asl = hv;
        }
        if (k0 + KSTEP < K) __builtin_prefetch(aptr + k0 + KSTEP, 0, 1);
#if USE_ASYNC_LDS
        asm volatile("s_wait_asynccnt 0x0" ::: "memory");
#endif
        __syncthreads();

        // A fragment (ISA 7.12.2, 16-bit A 16x32)
        v16h a;
        {
            const _Float16* pa = &As[(wr * 16 + mrow) * LDSW + hsel * 8];
            #pragma unroll
            for (int j = 0; j < 8; ++j) { a[j] = pa[j]; a[j + 8] = pa[j + 16]; }
        }
        #pragma unroll
        for (int t = 0; t < 4; ++t) {
            v16h b;
            const _Float16* pb = &Bs[(wc * 64 + t * 16 + mrow) * LDSW + hsel * 8];
            #pragma unroll
            for (int j = 0; j < 8; ++j) { b[j] = pb[j]; b[j + 8] = pb[j + 16]; }
            acc[t] = __builtin_amdgcn_wmma_f32_16x16x32_f16(
                false, a, false, b, (short)0, acc[t], false, false);
        }
        __syncthreads();
    }

    const int ncol = lane & 15;
    #pragma unroll
    for (int t = 0; t < 4; ++t) {
        const int gcol = colbase + wc * 64 + t * 16 + ncol;
        const float bv = bias[gcol];
        #pragma unroll
        for (int r = 0; r < 8; ++r) {
            const int orow = rowbase + wr * 16 + r + hsel * 8;
            float v = acc[t][r] + bv;
            if (EPI == 1)      v = fmaxf(v, 0.0f);
            else if (EPI == 2) v = 1.0f / (1.0f + __expf(-v));
            size_t off = (size_t)orow * NOUT + gcol;
            if (ADDY) Y[off] += v; else Y[off] = v;
        }
    }
}

// ---------------------------------------------------------------------------
// corr input transpose: (CIN, N) fp32 -> (N, KP) fp32 with zero K-padding
// 32x32 LDS tiles, coalesced both sides.
// ---------------------------------------------------------------------------
__global__ __launch_bounds__(256)
void transpose_pad_kernel(const float* __restrict__ src, float* __restrict__ dst,
                          int CIN, int N, int KP)
{
    __shared__ float t[32][33];
    const int nb = blockIdx.x * 32;
    const int kb = blockIdx.y * 32;
    const int tx = threadIdx.x;      // 0..31
    const int ty = threadIdx.y;      // 0..7
    #pragma unroll
    for (int j = ty; j < 32; j += 8) {
        int k = kb + j;
        t[j][tx] = (k < CIN) ? src[(size_t)k * N + nb + tx] : 0.0f;
    }
    __syncthreads();
    #pragma unroll
    for (int j = ty; j < 32; j += 8) {
        dst[(size_t)(nb + j) * KP + kb + tx] = t[tx][j];
    }
}

// fp32 -> f16 weight conversion, optional K padding (src ld = KSRC, dst ld = KP)
__global__ void cvt_f16_kernel(const float* __restrict__ s, _Float16* __restrict__ d, int n)
{
    int i = blockIdx.x * blockDim.x + threadIdx.x;
    if (i < n) d[i] = (_Float16)s[i];
}

__global__ void cvt_pad_f16_kernel(const float* __restrict__ s, _Float16* __restrict__ d,
                                   int rows, int KSRC, int KP)
{
    int i = blockIdx.x * blockDim.x + threadIdx.x;
    if (i >= rows * KP) return;
    int r = i / KP, k = i - r * KP;
    d[i] = (k < KSRC) ? (_Float16)s[(size_t)r * KSRC + k] : (_Float16)0.0f;
}

// ---------------------------------------------------------------------------
__device__ __forceinline__ float block_reduce384(float v, float* red, int tid)
{
    red[tid] = v;
    if (tid < 128) red[384 + tid] = 0.0f;
    __syncthreads();
    for (int off = 256; off > 0; off >>= 1) {
        if (tid < off) red[tid] += red[tid + off];
        __syncthreads();
    }
    float s = red[0];
    __syncthreads();
    return s;
}

__global__ __launch_bounds__(384)
void ln_kernel(const float* __restrict__ in, float* __restrict__ out,
               const float* __restrict__ g, const float* __restrict__ b, int D)
{
    __shared__ float red[512];
    int row = blockIdx.x, tid = threadIdx.x;
    float v = in[(size_t)row * D + tid];
    float mean = block_reduce384(v, red, tid) / D;
    float dlt = v - mean;
    float var = block_reduce384(dlt * dlt, red, tid) / D;
    out[(size_t)row * D + tid] = dlt * rsqrtf(var + 1e-3f) * g[tid] + b[tid];
}

__global__ __launch_bounds__(384)
void fuse_add_ln_kernel(const float* __restrict__ net, const float* __restrict__ inp,
                        const float* __restrict__ cc, float* __restrict__ out,
                        const float* __restrict__ g, const float* __restrict__ b,
                        int N, int D)
{
    __shared__ float red[512];
    int row = blockIdx.x, tid = threadIdx.x;
    float v = net[(size_t)tid * N + row] + inp[(size_t)tid * N + row]
            + cc[(size_t)row * D + tid];
    float mean = block_reduce384(v, red, tid) / D;
    float dlt = v - mean;
    float var = block_reduce384(dlt * dlt, red, tid) / D;
    out[(size_t)row * D + tid] = dlt * rsqrtf(var + 1e-3f) * g[tid] + b[tid];
}

__global__ void combine_kernel(const float* __restrict__ base, const float* __restrict__ gate,
                               const float* __restrict__ res, float* __restrict__ out, int n)
{
    int i = blockIdx.x * blockDim.x + threadIdx.x;
    if (i < n) out[i] = base[i] + gate[i] * res[i];
}

// ---------------------------------------------------------------------------
__global__ __launch_bounds__(256)
void neighbors_kernel(const int* __restrict__ kk, const int* __restrict__ jj,
                      int* __restrict__ ix, int* __restrict__ jx, int N)
{
    __shared__ unsigned smx[256];
    __shared__ unsigned smn[256];
    int r = blockIdx.x, tid = threadIdx.x;
    int kkr = kk[r], jjr = jj[r];
    unsigned bmax = 0u, bmin = 0xFFFFFFFFu;
    for (int j = tid; j < N; j += 256) {
        bool same = (kk[j] == kkr);
        int pv = (same && jj[j] < jjr) ? jj[j] : 0;
        int nv = (same && jj[j] > jjr) ? jj[j] : N;
        unsigned kx = ((unsigned)pv << 16) | (unsigned)(65535 - j);
        unsigned kn = ((unsigned)nv << 16) | (unsigned)j;
        if (kx > bmax) bmax = kx;
        if (kn < bmin) bmin = kn;
    }
    smx[tid] = bmax; smn[tid] = bmin;
    __syncthreads();
    for (int off = 128; off > 0; off >>= 1) {
        if (tid < off) {
            if (smx[tid + off] > smx[tid]) smx[tid] = smx[tid + off];
            if (smn[tid + off] < smn[tid]) smn[tid] = smn[tid + off];
        }
        __syncthreads();
    }
    if (tid == 0) {
        int iv = 65535 - (int)(smx[0] & 0xFFFFu);
        int jv = (int)(smn[0] & 0xFFFFu);
        ix[r] = iv < N ? iv : N - 1;
        jx[r] = jv < N ? jv : N - 1;
    }
}

// ---------------------------------------------------------------------------
__device__ __forceinline__ unsigned enc_f(float x) {
    unsigned u = __float_as_uint(x);
    return (u & 0x80000000u) ? ~u : (u | 0x80000000u);
}
__device__ __forceinline__ float dec_f(unsigned u) {
    return (u & 0x80000000u) ? __uint_as_float(u ^ 0x80000000u) : __uint_as_float(~u);
}

__global__ void seg_init_kernel(unsigned* __restrict__ menc, float* __restrict__ den,
                                float* __restrict__ num, int n)
{
    int i = blockIdx.x * blockDim.x + threadIdx.x;
    if (i < n) { menc[i] = 0x007FFFFFu; den[i] = 0.0f; num[i] = 0.0f; }
}

__global__ void seg_max_kernel(const float* __restrict__ gx, const int* __restrict__ map,
                               unsigned* __restrict__ menc, int n, int D)
{
    int i = blockIdx.x * blockDim.x + threadIdx.x;
    if (i >= n) return;
    int row = i / D, d = i - row * D;
    atomicMax(&menc[(size_t)map[row] * D + d], enc_f(gx[i]));
}

__global__ void seg_acc_kernel(const float* __restrict__ gx, const float* __restrict__ fx,
                               const int* __restrict__ map, const unsigned* __restrict__ menc,
                               float* __restrict__ den, float* __restrict__ num, int n, int D)
{
    int i = blockIdx.x * blockDim.x + threadIdx.x;
    if (i >= n) return;
    int row = i / D, d = i - row * D;
    size_t gi = (size_t)map[row] * D + d;
    float w = expf(gx[i] - dec_f(menc[gi]));
    atomicAdd(&den[gi], w);
    atomicAdd(&num[gi], w * fx[i]);
}

__global__ void seg_div_kernel(const float* __restrict__ num, const float* __restrict__ den,
                               float* __restrict__ y, int n)
{
    int i = blockIdx.x * blockDim.x + threadIdx.x;
    if (i < n) y[i] = (den[i] > 0.0f) ? num[i] / den[i] : 0.0f;
}

// ---------------------------------------------------------------------------
__global__ __launch_bounds__(256)
void heads_kernel(const float* __restrict__ x,
                  const float* __restrict__ wd, const float* __restrict__ bd,
                  const float* __restrict__ ww, const float* __restrict__ bw,
                  float* __restrict__ dout, float* __restrict__ wout, int N, int D)
{
    int wid = threadIdx.x >> 5, lane = threadIdx.x & 31;
    int row = blockIdx.x * 8 + wid;
    if (row >= N) return;
    const float* xr = x + (size_t)row * D;
    float a0 = 0.f, a1 = 0.f, a2 = 0.f, a3 = 0.f;
    for (int k = lane; k < D; k += 32) {
        float v = fmaxf(xr[k], 0.0f);
        a0 += v * wd[k]; a1 += v * wd[D + k];
        a2 += v * ww[k]; a3 += v * ww[D + k];
    }
    for (int off = 16; off > 0; off >>= 1) {
        a0 += __shfl_down(a0, off); a1 += __shfl_down(a1, off);
        a2 += __shfl_down(a2, off); a3 += __shfl_down(a3, off);
    }
    if (lane == 0) {
        dout[(size_t)row * 2 + 0] = a0 + bd[0];
        dout[(size_t)row * 2 + 1] = a1 + bd[1];
        wout[(size_t)row * 2 + 0] = 1.0f / (1.0f + __expf(-(a2 + bw[0])));
        wout[(size_t)row * 2 + 1] = 1.0f / (1.0f + __expf(-(a3 + bw[1])));
    }
}

// ===========================================================================
// host launcher
// ===========================================================================
extern "C" void kernel_launch(void* const* d_in, const int* in_sizes, int n_in,
                              void* d_out, int out_size, void* d_ws, size_t ws_size,
                              hipStream_t stream)
{
    constexpr int DIM = 384;
    constexpr int GKK = 768;
    constexpr int GIJ = 64;
    const int N   = in_sizes[4];
    const int CIN = in_sizes[2] / N;          // 882
    const int KP  = ((CIN + KSTEP - 1) / KSTEP) * KSTEP;   // 896

    const float* net  = (const float*)d_in[0];
    const float* inp  = (const float*)d_in[1];
    const float* corr = (const float*)d_in[2];
    const int*   jjp  = (const int*)d_in[4];
    const int*   kkp  = (const int*)d_in[5];
    const int*   kkmap= (const int*)d_in[6];
    const int*   ijmap= (const int*)d_in[8];
#define PF(i) ((const float*)d_in[(i)])
    // sorted-pytree leaf order:
    // agg_ij:10..15  agg_kk:16..21  c1:22..25  c2:26..29
    // corr l1:30,31 l2:32,33 l3:34,35 ln:36(b),37(g)
    // d:38,39  gr1:40..45  gr2:46..51  gru_ln1:52,53 gru_ln2:54,55
    // norm:56,57  w:58,59

    char* wsp = (char*)d_ws;
    auto carve = [&](size_t bytes) -> void* {
        void* r = (void*)wsp; wsp += (bytes + 255) & ~(size_t)255; return r;
    };
    const size_t ACTB = (size_t)N * DIM * sizeof(float);
    float* xb = (float*)carve(ACTB);
    float* t0 = (float*)carve(ACTB);
    float* t1 = (float*)carve(ACTB);
    float* t2 = (float*)carve(ACTB);
    float* t3 = (float*)carve(ACTB);
    float* corrT = (float*)carve((size_t)N * KP * sizeof(float));
    unsigned* menc = (unsigned*)carve((size_t)GKK * DIM * 4);
    float* den  = (float*)carve((size_t)GKK * DIM * 4);
    float* num  = (float*)carve((size_t)GKK * DIM * 4);
    float* yseg = (float*)carve((size_t)GKK * DIM * 4);
    int* ixb = (int*)carve((size_t)N * 4);
    int* jxb = (int*)carve((size_t)N * 4);

    const size_t WSQ = (size_t)DIM * DIM;
    _Float16* h_l1p = (_Float16*)carve((size_t)DIM * KP * 2);   // padded corr.l1
    _Float16* hw[18];
    for (int i = 0; i < 18; ++i) hw[i] = (_Float16*)carve(WSQ * 2);
    _Float16 *h_l2=hw[0], *h_l3=hw[1],
             *h_c1w0=hw[2], *h_c1w1=hw[3], *h_c2w0=hw[4], *h_c2w1=hw[5],
             *h_kkf=hw[6], *h_kkg=hw[7], *h_kkh=hw[8],
             *h_ijf=hw[9], *h_ijg=hw[10], *h_ijh=hw[11],
             *h_g1gate=hw[12], *h_g1r1=hw[13], *h_g1r2=hw[14],
             *h_g2gate=hw[15], *h_g2r1=hw[16], *h_g2r2=hw[17];

    auto cvt = [&](_Float16* dst, const float* src, int n) {
        cvt_f16_kernel<<<(n + 255) / 256, 256, 0, stream>>>(src, dst, n);
    };
    {
        int np = DIM * KP;
        cvt_pad_f16_kernel<<<(np + 255) / 256, 256, 0, stream>>>(PF(31), h_l1p, DIM, CIN, KP);
    }
    cvt(h_l2, PF(33), (int)WSQ);   cvt(h_l3, PF(35), (int)WSQ);
    cvt(h_c1w0, PF(23), (int)WSQ); cvt(h_c1w1, PF(25), (int)WSQ);
    cvt(h_c2w0, PF(27), (int)WSQ); cvt(h_c2w1, PF(29), (int)WSQ);
    cvt(h_kkf, PF(17), (int)WSQ);  cvt(h_kkg, PF(19), (int)WSQ);
    cvt(h_kkh, PF(21), (int)WSQ);
    cvt(h_ijf, PF(11), (int)WSQ);  cvt(h_ijg, PF(13), (int)WSQ);
    cvt(h_ijh, PF(15), (int)WSQ);
    cvt(h_g1gate, PF(41), (int)WSQ); cvt(h_g1r1, PF(43), (int)WSQ);
    cvt(h_g1r2, PF(45), (int)WSQ);
    cvt(h_g2gate, PF(47), (int)WSQ); cvt(h_g2r1, PF(49), (int)WSQ);
    cvt(h_g2r2, PF(51), (int)WSQ);

    // transpose + pad corr: (CIN,N) -> (N,KP)
    {
        dim3 g(N / 32, KP / 32), b(32, 8);
        transpose_pad_kernel<<<g, b, 0, stream>>>(corr, corrT, CIN, N, KP);
    }

    // WMMA GEMM dispatch (M=N rows, Nout=384, runtime K % 32 == 0)
    const dim3 fg(N / TILE_M, (DIM + TILE_N - 1) / TILE_N);
    auto gemmF = [&](const float* X, const _Float16* W, const float* bias, float* Y,
                     int K, const int* gth, int addY, int epi, int reluX) {
        if (gth && addY)
            gemm_wmma_fast<0, true,  true,  false><<<fg, 256, 0, stream>>>(X, W, bias, Y, N, K, gth);
        else if (gth)
            gemm_wmma_fast<1, true,  false, false><<<fg, 256, 0, stream>>>(X, W, bias, Y, N, K, gth);
        else if (addY)
            gemm_wmma_fast<0, false, true,  false><<<fg, 256, 0, stream>>>(X, W, bias, Y, N, K, nullptr);
        else if (reluX)
            gemm_wmma_fast<0, false, false, true ><<<fg, 256, 0, stream>>>(X, W, bias, Y, N, K, nullptr);
        else if (epi == 1)
            gemm_wmma_fast<1, false, false, false><<<fg, 256, 0, stream>>>(X, W, bias, Y, N, K, nullptr);
        else if (epi == 2)
            gemm_wmma_fast<2, false, false, false><<<fg, 256, 0, stream>>>(X, W, bias, Y, N, K, nullptr);
        else
            gemm_wmma_fast<0, false, false, false><<<fg, 256, 0, stream>>>(X, W, bias, Y, N, K, nullptr);
    };

    // neighbors
    neighbors_kernel<<<N, 256, 0, stream>>>(kkp, jjp, ixb, jxb, N);

    // corr MLP: relu(l1(c)) -> ln(l2(.)) -> l3(relu(.))
    gemmF(corrT, h_l1p, PF(30), t0, KP, nullptr, 0, 1, 0);
    gemmF(t0, h_l2, PF(32), t1, DIM, nullptr, 0, 0, 0);
    ln_kernel<<<N, 384, 0, stream>>>(t1, t1, PF(37), PF(36), DIM);
    gemmF(t1, h_l3, PF(34), t2, DIM, nullptr, 0, 0, 1);
    fuse_add_ln_kernel<<<N, 384, 0, stream>>>(net, inp, t2, xb, PF(57), PF(56), N, DIM);

    // x += mlp2(c1, x[ix]); x += mlp2(c2, x[jx])
    gemmF(xb, h_c1w0, PF(22), t0, DIM, ixb, 0, 1, 0);
    gemmF(t0, h_c1w1, PF(24), xb, DIM, nullptr, 1, 0, 0);
    gemmF(xb, h_c2w0, PF(26), t0, DIM, jxb, 0, 1, 0);
    gemmF(t0, h_c2w1, PF(28), xb, DIM, nullptr, 1, 0, 0);

    // soft_agg kk (G=768)
    gemmF(xb, h_kkf, PF(16), t0, DIM, nullptr, 0, 0, 0);
    gemmF(xb, h_kkg, PF(18), t1, DIM, nullptr, 0, 0, 0);
    {
        int ns = GKK * DIM, ne = N * DIM;
        seg_init_kernel<<<(ns + 255) / 256, 256, 0, stream>>>(menc, den, num, ns);
        seg_max_kernel <<<(ne + 255) / 256, 256, 0, stream>>>(t1, kkmap, menc, ne, DIM);
        seg_acc_kernel <<<(ne + 255) / 256, 256, 0, stream>>>(t1, t0, kkmap, menc, den, num, ne, DIM);
        seg_div_kernel <<<(ns + 255) / 256, 256, 0, stream>>>(num, den, yseg, ns);
    }
    gemmF(yseg, h_kkh, PF(20), xb, DIM, kkmap, 1, 0, 0);

    // soft_agg ij (G=64)
    gemmF(xb, h_ijf, PF(10), t0, DIM, nullptr, 0, 0, 0);
    gemmF(xb, h_ijg, PF(12), t1, DIM, nullptr, 0, 0, 0);
    {
        int ns = GIJ * DIM, ne = N * DIM;
        seg_init_kernel<<<(ns + 255) / 256, 256, 0, stream>>>(menc, den, num, ns);
        seg_max_kernel <<<(ne + 255) / 256, 256, 0, stream>>>(t1, ijmap, menc, ne, DIM);
        seg_acc_kernel <<<(ne + 255) / 256, 256, 0, stream>>>(t1, t0, ijmap, menc, den, num, ne, DIM);
        seg_div_kernel <<<(ns + 255) / 256, 256, 0, stream>>>(num, den, yseg, ns);
    }
    gemmF(yseg, h_ijh, PF(14), xb, DIM, ijmap, 1, 0, 0);

    // gated residual 1
    ln_kernel<<<N, 384, 0, stream>>>(xb, t2, PF(53), PF(52), DIM);
    gemmF(t2, h_g1gate, PF(40), t0, DIM, nullptr, 0, 2, 0);
    gemmF(t2, h_g1r1,   PF(42), t1, DIM, nullptr, 0, 1, 0);
    gemmF(t1, h_g1r2,   PF(44), t3, DIM, nullptr, 0, 0, 0);
    combine_kernel<<<(N * DIM + 255) / 256, 256, 0, stream>>>(t2, t0, t3, xb, N * DIM);

    // gated residual 2 -> output x
    float* outx = (float*)d_out;
    float* outd = outx + (size_t)N * DIM;
    float* outw = outd + (size_t)N * 2;
    ln_kernel<<<N, 384, 0, stream>>>(xb, t2, PF(55), PF(54), DIM);
    gemmF(t2, h_g2gate, PF(46), t0, DIM, nullptr, 0, 2, 0);
    gemmF(t2, h_g2r1,   PF(48), t1, DIM, nullptr, 0, 1, 0);
    gemmF(t1, h_g2r2,   PF(50), t3, DIM, nullptr, 0, 0, 0);
    combine_kernel<<<(N * DIM + 255) / 256, 256, 0, stream>>>(t2, t0, t3, outx, N * DIM);

    heads_kernel<<<(N + 7) / 8, 256, 0, stream>>>(outx, PF(39), PF(38), PF(59), PF(58),
                                                  outd, outw, N, DIM);
#undef PF
    (void)n_in; (void)out_size; (void)ws_size;
}
